// LocalRelationalLayer_21457656611277
// MI455X (gfx1250) — matrix operand order repaired and checked by reading, not compiled, not run on hardware
//
#include <hip/hip_runtime.h>
#include <hip/hip_bf16.h>

// ---------------- types for WMMA ----------------
typedef __bf16 bf16;
typedef __attribute__((ext_vector_type(16))) __bf16 v16bf;
typedef __attribute__((ext_vector_type(8)))  __bf16 v8bf;
typedef __attribute__((ext_vector_type(8)))  float  v8f;

// ---------------- problem constants ----------------
#define BB   2
#define CC   256
#define HH   56
#define WW   56
#define LL   (HH*WW)          // 3136
#define CMC  32
#define MG   8
#define KW   7
#define KK   49
#define PADP 3
#define HP   62               // H + 2*PAD
#define MIDW 16

// LDS tile geometry for the GEMM kernels
#define LDSTR 272             // padded row stride (elements) -> 32B-aligned frag loads
#define SMEM_BYTES (4 * 64 * LDSTR * 2)   // Ah, Al, Xh, Xl  (bf16) = 139264 B

static __device__ __forceinline__ v8f wmma_bf16(v16bf a, v16bf b, v8f c) {
  // (neg_a, A, neg_b, B, c_mod, C, reuse_a, reuse_b)
  return __builtin_amdgcn_wmma_f32_16x16x32_bf16(false, a, false, b, (short)0, c, false, false);
}

// A fragment: 16x32 bf16.  lane: M = lane&15, h = lane>>4.
// element e<8  -> K = 32*kk + 8h + e        (contiguous 8)
// element e>=8 -> K = 32*kk + 16 + 8h + e-8 (contiguous 8)
static __device__ __forceinline__ v16bf ldsA_frag(const bf16* A, int mt, int kk, int lane) {
  int m = mt * 16 + (lane & 15);
  int h = lane >> 4;
  const bf16* p = A + m * LDSTR + kk * 32;
  v8bf a0 = *(const v8bf*)(p + 8 * h);
  v8bf a1 = *(const v8bf*)(p + 16 + 8 * h);
  return __builtin_shufflevector(a0, a1, 0,1,2,3,4,5,6,7,8,9,10,11,12,13,14,15);
}

// B fragment: 32x16 bf16.  lane: N = lane&15, h = lane>>4; e<8 -> K = 32*kk + 8h + e, e>=8 -> K = 32*kk + 16 + 8h + e-8.
// X stored transposed in LDS as [n][c] so the 16 K-values are two v8bf loads.
static __device__ __forceinline__ v16bf ldsB_frag(const bf16* X, int nsub, int kk, int lane) {
  int n = nsub * 16 + (lane & 15);
  int h = lane >> 4;
  const bf16* p = X + n * LDSTR + kk * 32;
  v8bf b0 = *(const v8bf*)(p + 8 * h);
  v8bf b1 = *(const v8bf*)(p + 16 + 8 * h);
  return __builtin_shufflevector(b0, b1, 0,1,2,3,4,5,6,7,8,9,10,11,12,13,14,15);
}

// 64(M) x 64(N) x 256(K) core: wave w owns N-subtile w (16 cols), all 4 M-tiles.
// bf16x3 split precision: acc += Ah*Bh + Ah*Bl + Al*Bh  (96 WMMAs / wave).
static __device__ __forceinline__ void gemm64_core(const bf16* Ah, const bf16* Al,
                                                   const bf16* Xh, const bf16* Xl,
                                                   int wave, int lane, v8f acc[4]) {
  #pragma unroll
  for (int kk = 0; kk < 8; ++kk) {
    v16bf bh = ldsB_frag(Xh, wave, kk, lane);
    v16bf bl = ldsB_frag(Xl, wave, kk, lane);
    #pragma unroll
    for (int mt = 0; mt < 4; ++mt) {
      v16bf ah = ldsA_frag(Ah, mt, kk, lane);
      v16bf al = ldsA_frag(Al, mt, kk, lane);
      acc[mt] = wmma_bf16(ah, bh, acc[mt]);
      acc[mt] = wmma_bf16(ah, bl, acc[mt]);
      acc[mt] = wmma_bf16(al, bh, acc[mt]);
    }
  }
}

static __device__ __forceinline__ void split_store(bf16* H, bf16* L, int idx, float v) {
  bf16 h = (bf16)v;
  H[idx] = h;
  L[idx] = (bf16)(v - (float)h);
}

// ---------------- kernel 0: geometry prior  gpk[32][7][7] ----------------
__global__ void k_gpk(const float* __restrict__ gp_w1, const float* __restrict__ gp_b1,
                      const float* __restrict__ gp_w2, const float* __restrict__ gp_b2,
                      float* __restrict__ gpk) {
  for (int idx = threadIdx.x; idx < CMC * KK; idx += blockDim.x) {
    int cm = idx / KK, r = idx % KK, i = r / KW, j = r % KW;
    float xp = (float)(j - 3);     // x_pos
    float yp = (float)(3 - i);     // y_pos (reversed row order)
    float acc = gp_b2[cm];
    #pragma unroll
    for (int m = 0; m < MIDW; ++m) {
      float h1 = gp_w1[m * 2 + 0] * xp + gp_w1[m * 2 + 1] * yp + gp_b1[m];
      h1 = fmaxf(h1, 0.0f);
      acc += gp_w2[cm * MIDW + m] * h1;
    }
    gpk[idx] = acc;
  }
}

// ---------------- kernel 1: fill km_pad with bias (border stays = k_b) ----------------
__global__ void k_fill_kmpad(const float* __restrict__ k_b, float* __restrict__ km_pad) {
  int idx = blockIdx.x * blockDim.x + threadIdx.x;
  if (idx < BB * CMC * HP * HP) {
    int o = (idx / (HP * HP)) % CMC;
    km_pad[idx] = k_b[o];
  }
}

// ---------------- kernel 2: k/q projection GEMM  [64x256] x [256 x 64-tile] ----------------
__global__ void __launch_bounds__(128)
k_gemm_kq(const float* __restrict__ xin,
          const float* __restrict__ k_w, const float* __restrict__ k_b,
          const float* __restrict__ q_w, const float* __restrict__ q_b,
          float* __restrict__ km_pad, float* __restrict__ qm) {
  extern __shared__ __align__(64) char smem[];
  bf16* Ah = (bf16*)smem;
  bf16* Al = Ah + 64 * LDSTR;
  bf16* Xh = Al + 64 * LDSTR;
  bf16* Xl = Xh + 64 * LDSTR;

  int b  = blockIdx.x / 49;
  int n0 = (blockIdx.x % 49) * 64;
  int tid = threadIdx.x;

  // stage A = concat(k_w, q_w) rows 0..63, split to bf16 hi/lo
  for (int idx = tid; idx < 64 * 256; idx += 128) {
    int row = idx >> 8, col = idx & 255;
    float w = (row < 32) ? k_w[row * 256 + col] : q_w[(row - 32) * 256 + col];
    split_store(Ah, Al, row * LDSTR + col, w);
  }
  // stage X tile (transposed to [n][c]); coalesced global reads along n
  const float* xb = xin + (size_t)b * CC * LL + n0;
  for (int idx = tid; idx < 64 * 256; idx += 128) {
    int n = idx & 63, c = idx >> 6;
    split_store(Xh, Xl, n * LDSTR + c, xb[c * LL + n]);
  }
  __syncthreads();

  int wave = tid >> 5, lane = tid & 31;
  v8f z = {0.f,0.f,0.f,0.f,0.f,0.f,0.f,0.f};
  v8f acc[4] = {z, z, z, z};
  gemm64_core(Ah, Al, Xh, Xl, wave, lane, acc);

  int nloc = n0 + wave * 16 + (lane & 15);
  int y = nloc / WW, xx = nloc % WW;
  int h = lane >> 4;
  #pragma unroll
  for (int mt = 0; mt < 4; ++mt) {
    #pragma unroll
    for (int r = 0; r < 8; ++r) {
      int o = mt * 16 + 8 * h + r;
      float v = acc[mt][r];
      if (o < CMC) {
        km_pad[((b * CMC + o) * HP + (y + PADP)) * HP + (xx + PADP)] = v + k_b[o];
      } else {
        qm[(b * CMC + (o - CMC)) * LL + nloc] = v + q_b[o - CMC];
      }
    }
  }
}

// ---------------- kernel 3: local-window attention (VALU + v_exp_f32) ----------------
__global__ void __launch_bounds__(256)
k_attn(const float* __restrict__ km_pad, const float* __restrict__ qm,
       const float* __restrict__ gpk, const float* __restrict__ xin,
       float* __restrict__ pre) {
  int idx = blockIdx.x * blockDim.x + threadIdx.x;   // over B*CM*L, l fastest
  if (idx >= BB * CMC * LL) return;
  int l  = idx % LL;
  int cm = (idx / LL) % CMC;
  int b  = idx / (CMC * LL);
  int y = l / WW, x = l % WW;

  float q = qm[(b * CMC + cm) * LL + l];
  const float* kmrow = km_pad + ((size_t)(b * CMC + cm) * HP) * HP;
  const float* gpr   = gpk + cm * KK;

  float ck[KK];
  #pragma unroll
  for (int i = 0; i < KW; ++i) {           // softmax over each width-7 row
    float v[KW];
    float mx = -3.4e38f;
    #pragma unroll
    for (int j = 0; j < KW; ++j) {
      float a = kmrow[(y + i) * HP + (x + j)] * q + gpr[i * KW + j];
      v[j] = a;
      mx = fmaxf(mx, a);
    }
    float s = 0.f;
    #pragma unroll
    for (int j = 0; j < KW; ++j) { float e = __expf(v[j] - mx); ck[i * KW + j] = e; s += e; }
    float inv = 1.0f / s;
    #pragma unroll
    for (int j = 0; j < KW; ++j) ck[i * KW + j] *= inv;
  }

  for (int m = 0; m < MG; ++m) {
    int ch = m * CMC + cm;
    const float* xc = xin + ((size_t)(b * CC + ch)) * LL;
    float acc = 0.f;
    #pragma unroll
    for (int i = 0; i < KW; ++i) {
      int yy = y + i - PADP;
      #pragma unroll
      for (int j = 0; j < KW; ++j) {
        int xx = x + j - PADP;
        float xv = ((unsigned)yy < (unsigned)HH && (unsigned)xx < (unsigned)WW)
                   ? xc[yy * WW + xx] : 0.f;
        acc += ck[i * KW + j] * xv;
      }
    }
    pre[(size_t)(b * CC + ch) * LL + l] = acc;
  }
}

// ---------------- kernel 4: final 1x1 conv GEMM  [256x256] x [256 x 6272] ----------------
__global__ void __launch_bounds__(128)
k_gemm_out(const float* __restrict__ pre,
           const float* __restrict__ f_w, const float* __restrict__ f_b,
           float* __restrict__ out) {
  extern __shared__ __align__(64) char smem[];
  bf16* Ah = (bf16*)smem;
  bf16* Al = Ah + 64 * LDSTR;
  bf16* Xh = Al + 64 * LDSTR;
  bf16* Xl = Xh + 64 * LDSTR;

  int nb = blockIdx.x % 98;          // N tiles (B*49)
  int mb = blockIdx.x / 98;          // M quarter (0..3) -> rows mb*64..+63
  int b  = nb / 49;
  int n0 = (nb % 49) * 64;
  int tid = threadIdx.x;

  for (int idx = tid; idx < 64 * 256; idx += 128) {
    int row = idx >> 8, col = idx & 255;
    split_store(Ah, Al, row * LDSTR + col, f_w[(mb * 64 + row) * 256 + col]);
  }
  const float* pb = pre + (size_t)b * CC * LL + n0;
  for (int idx = tid; idx < 64 * 256; idx += 128) {
    int n = idx & 63, c = idx >> 6;
    split_store(Xh, Xl, n * LDSTR + c, pb[c * LL + n]);
  }
  __syncthreads();

  int wave = tid >> 5, lane = tid & 31;
  v8f z = {0.f,0.f,0.f,0.f,0.f,0.f,0.f,0.f};
  v8f acc[4] = {z, z, z, z};
  gemm64_core(Ah, Al, Xh, Xl, wave, lane, acc);

  int nloc = n0 + wave * 16 + (lane & 15);
  int h = lane >> 4;
  #pragma unroll
  for (int mt = 0; mt < 4; ++mt) {
    #pragma unroll
    for (int r = 0; r < 8; ++r) {
      int o = mb * 64 + mt * 16 + 8 * h + r;
      out[(size_t)(b * CC + o) * LL + nloc] = acc[mt][r] + f_b[o];
    }
  }
}

// ---------------- launcher ----------------
extern "C" void kernel_launch(void* const* d_in, const int* in_sizes, int n_in,
                              void* d_out, int out_size, void* d_ws, size_t ws_size,
                              hipStream_t stream) {
  const float* x     = (const float*)d_in[0];
  const float* k_w   = (const float*)d_in[1];
  const float* k_b   = (const float*)d_in[2];
  const float* q_w   = (const float*)d_in[3];
  const float* q_b   = (const float*)d_in[4];
  const float* gp_w1 = (const float*)d_in[5];
  const float* gp_b1 = (const float*)d_in[6];
  const float* gp_w2 = (const float*)d_in[7];
  const float* gp_b2 = (const float*)d_in[8];
  const float* f_w   = (const float*)d_in[9];
  const float* f_b   = (const float*)d_in[10];
  float* out = (float*)d_out;

  // workspace partition (floats): gpk | km_pad | qm | pre   (~8.2 MB total)
  float* gpk    = (float*)d_ws;
  float* km_pad = gpk + 2048;                       // 1568 used
  float* qm     = km_pad + BB * CMC * HP * HP;      // 246016
  float* pre    = qm + BB * CMC * LL;               // 200704; pre = 1605632

  k_gpk       <<<1,   64, 0, stream>>>(gp_w1, gp_b1, gp_w2, gp_b2, gpk);
  k_fill_kmpad<<<961, 256, 0, stream>>>(k_b, km_pad);
  k_gemm_kq   <<<98,  128, SMEM_BYTES, stream>>>(x, k_w, k_b, q_w, q_b, km_pad, qm);
  k_attn      <<<784, 256, 0, stream>>>(km_pad, qm, gpk, x, pre);
  k_gemm_out  <<<392, 128, SMEM_BYTES, stream>>>(pre, f_w, f_b, out);
}
